// RGCN4_31610959299138
// MI455X (gfx1250) — matrix operand order repaired and checked by reading, not compiled, not run on hardware
//
#include <hip/hip_runtime.h>
#include <hip/hip_bf16.h>

typedef _Float16 half_t;
typedef __attribute__((ext_vector_type(16))) _Float16 v16h;
typedef __attribute__((ext_vector_type(8)))  _Float16 v8h;
typedef __attribute__((ext_vector_type(8)))  float    v8f;
typedef __attribute__((ext_vector_type(4)))  unsigned int u32x4;
typedef __attribute__((ext_vector_type(8)))  int      i32x8;
typedef __attribute__((ext_vector_type(4)))  int      i32x4;

#define HID 128
#define HEADS 2
#define CHAN 64

#if defined(__has_builtin)
#if __has_builtin(__builtin_amdgcn_tensor_load_to_lds) && __has_builtin(__builtin_amdgcn_s_wait_tensorcnt)
#define USE_TDM 1
#endif
#endif

// ---------------------------------------------------------------------------
// utility fills / converts
// ---------------------------------------------------------------------------
__global__ void fill_f32(float* p, float v, int n) {
    int i = blockIdx.x * blockDim.x + threadIdx.x;
    if (i < n) p[i] = v;
}
__global__ void fill_i32(int* p, int v, int n) {
    int i = blockIdx.x * blockDim.x + threadIdx.x;
    if (i < n) p[i] = v;
}
__global__ void f32_to_f16(const float* __restrict__ src, half_t* __restrict__ dst, int n) {
    int i = blockIdx.x * blockDim.x + threadIdx.x;
    if (i < n) dst[i] = (half_t)src[i];
}

// ---------------------------------------------------------------------------
// WMMA GEMM:  C[M x NC](f32) = A[M x K](f16, lda=K) @ B[K x NC](f16)
// Block: 256 threads = 8 waves; block tile = 64 rows x 128 cols:
//   wave w -> 16-col slice w; 4 row tiles of 16, B fragments held in registers
//   across row tiles. A slab (64*K f16, contiguous) staged via TDM when whole.
// ---------------------------------------------------------------------------
template <int K, int NC>
__global__ __launch_bounds__(256) void gemm_f16_wmma(
    const half_t* __restrict__ A, const half_t* __restrict__ B,
    float* __restrict__ Cout, int M)
{
    constexpr int NCP  = (NC + 15) & ~15;         // padded cols, <= 128
    constexpr int RT   = 4;                       // row tiles per block
    constexpr int ROWS = 16 * RT;
    constexpr int KC   = K / 32;                  // k-chunks
    __shared__ half_t Al[ROWS * K];               // row-major A slab
    __shared__ half_t Bt[NCP * K];                // B transposed: Bt[col*K + k]

    const int tid  = threadIdx.x;
    const int wave = tid >> 5;
    const int lane = tid & 31;
    const int rowBase = blockIdx.x * ROWS;

    // ---- stage B transposed (coalesced global read, LDS scatter) ----
    for (int i = tid; i < K * NC; i += 256) {
        int r = i / NC, c = i - r * NC;
        Bt[c * K + r] = B[i];
    }
    if constexpr (NCP > NC) {
        for (int i = tid; i < (NCP - NC) * K; i += 256) {
            int c = NC + i / K, r = i - (i / K) * K;
            Bt[c * K + r] = (half_t)0.f;
        }
    }

    // ---- stage A slab (64 x K, contiguous in global since lda == K) ----
    bool tdm_done = false;
#if defined(USE_TDM)
    if (rowBase + ROWS <= M) {
        if (wave == 0) {
            // 2-D TDM descriptor: data_size=2B, tile K x ROWS, stride K (uniform -> SGPRs)
            unsigned lds_a = (unsigned)(unsigned long long)(const void*)&Al[0]; // LDS byte address from low 32 bits
            unsigned long long ga = (unsigned long long)(const void*)(A + (size_t)rowBase * K);
            u32x4 g0;
            g0[0] = 1u;                                       // count=1, user mode, no gather
            g0[1] = lds_a;                                    // lds_addr (bytes)
            g0[2] = (unsigned)(ga & 0xffffffffu);             // global_addr[31:0]
            g0[3] = (unsigned)((ga >> 32) & 0x01ffffffu) | (2u << 30); // addr[56:32], type=2
            i32x8 g1;
            g1[0] = (int)(1u << 16);                          // data_size=1 (2 bytes), mask=0
            g1[1] = (int)(((unsigned)K & 0xffffu) << 16);     // tensor_dim0[15:0]
            g1[2] = (int)((((unsigned)K >> 16) & 0xffffu) |
                          (((unsigned)M & 0xffffu) << 16));   // td0 hi | tensor_dim1 lo
            g1[3] = (int)((((unsigned)M >> 16) & 0xffffu) |
                          (((unsigned)K) << 16));             // td1 hi | tile_dim0=K
            g1[4] = ROWS;                                     // tile_dim1=64, tile_dim2=0
            g1[5] = K;                                        // tensor_dim0_stride[31:0]
            g1[6] = 0;
            g1[7] = 0;
            i32x4 z4 = {0, 0, 0, 0};
#if __clang_major__ >= 23
            i32x8 z8 = {0, 0, 0, 0, 0, 0, 0, 0};
            __builtin_amdgcn_tensor_load_to_lds(g0, g1, z4, z4, z8, 0);
#else
            __builtin_amdgcn_tensor_load_to_lds(g0, g1, z4, z4, 0);
#endif
            __builtin_amdgcn_s_wait_tensorcnt(0);
        }
        tdm_done = true;
    }
#endif
    if (!tdm_done) {
        for (int i = tid; i < ROWS * K; i += 256) {
            int r = i / K, c = i - (i / K) * K;
            int gr = rowBase + r;
            Al[i] = (gr < M) ? A[(size_t)gr * K + c] : (half_t)0.f;
        }
    }
    __syncthreads();

    const int colBase = wave * 16;
    if (colBase >= NCP) return;

    const int ln15 = lane & 15;   // A row within tile; B/C/D column within tile
    const int hi   = lane >> 4;
    const int gc   = colBase + ln15;

    // ---- preload all B fragments (reused across the 4 row tiles) ----
    v16h bfrag[KC];
#pragma unroll
    for (int kc = 0; kc < KC; ++kc)
        bfrag[kc] = *(const v16h*)&Bt[gc * K + kc * 32 + hi * 16];

#pragma unroll
    for (int rt = 0; rt < RT; ++rt) {
        v8f acc = {};
#pragma unroll
        for (int kc = 0; kc < KC; ++kc) {
            // A 16x32 fragment = two contiguous 8-half runs of one row (2x ds_load_b128)
            const half_t* ap = &Al[(rt * 16 + ln15) * K + kc * 32 + hi * 16];
            v8h alo = *(const v8h*)ap;
            v8h ahi = *(const v8h*)(ap + 8);
            v16h a = __builtin_shufflevector(alo, ahi,
                     0, 1, 2, 3, 4, 5, 6, 7, 8, 9, 10, 11, 12, 13, 14, 15);
            acc = __builtin_amdgcn_wmma_f32_16x16x32_f16(
                false, a, false, bfrag[kc], (short)0, acc, false, false);
        }

        // D 16x16 f32: lane column = gc, VGPR v -> row rt*16 + v + hi*8
        const int gr0 = rowBase + rt * 16;
        float* cptr = Cout + (size_t)(gr0 + hi * 8) * NC + gc;
        const bool fastpath = (gr0 + 16 <= M) && (gc < NC);
        if (fastpath) {
#pragma unroll
            for (int v = 0; v < 8; ++v) cptr[v * NC] = acc[v];   // immediate offsets
        } else {
#pragma unroll
            for (int v = 0; v < 8; ++v) {
                int gr = gr0 + v + hi * 8;
                if (gr < M && gc < NC) Cout[(size_t)gr * NC + gc] = acc[v];
            }
        }
    }
}

// ---------------------------------------------------------------------------
// BatchNorm machinery (batch statistics, eps = 1e-5)
// ---------------------------------------------------------------------------
__global__ void col_stats(const float* __restrict__ X, float* sum, float* sumsq, int n) {
    int c = threadIdx.x;                      // 128 threads = one column each
    float s = 0.f, q = 0.f;
    for (int r = blockIdx.x; r < n; r += gridDim.x) {
        __builtin_prefetch(&X[(size_t)(r + gridDim.x) * HID + c], 0, 0);
        float v = X[(size_t)r * HID + c];
        s += v; q += v * v;
    }
    atomicAdd(&sum[c], s);
    atomicAdd(&sumsq[c], q);
}
__global__ void finalize_stats(const float* sum, const float* sumsq,
                               const float* gamma, const float* beta,
                               float* scale, float* shift, int n) {
    int c = threadIdx.x;
    if (c >= HID) return;
    float inv_n = 1.f / (float)n;
    float mu  = sum[c] * inv_n;
    float var = sumsq[c] * inv_n - mu * mu;
    float sc  = gamma[c] * rsqrtf(var + 1e-5f);
    scale[c] = sc;
    shift[c] = beta[c] - mu * sc;
}
__global__ void bn_relu_to_f16(const float* __restrict__ X,
                               const float* __restrict__ scale,
                               const float* __restrict__ shift,
                               half_t* __restrict__ out, int total) {
    int i = blockIdx.x * blockDim.x + threadIdx.x;
    if (i >= total) return;
    int c = i & (HID - 1);
    float v = X[i] * scale[c] + shift[c];
    out[i] = (half_t)fmaxf(v, 0.f);
}

// ---------------------------------------------------------------------------
// GAT pieces
// ---------------------------------------------------------------------------
__global__ void gat_alphas(const float* __restrict__ xp,
                           const float* __restrict__ a_s, const float* __restrict__ a_d,
                           float* __restrict__ asrc, float* __restrict__ adst, int n) {
    int i = blockIdx.x * blockDim.x + threadIdx.x;
    if (i >= n) return;
    const float* row = xp + (size_t)i * HID;
#pragma unroll
    for (int h = 0; h < HEADS; ++h) {
        float s = 0.f, d = 0.f;
        for (int c = 0; c < CHAN; ++c) {
            float v = row[h * CHAN + c];
            s += v * a_s[h * CHAN + c];
            d += v * a_d[h * CHAN + c];
        }
        asrc[i * HEADS + h] = s;
        adst[i * HEADS + h] = d;
    }
}

__device__ __forceinline__ int f2ord(float v) {
    int o = __float_as_int(v);
    return (o >= 0) ? o : (o ^ 0x7fffffff);
}
__device__ __forceinline__ float ord2f(int o) {
    return __int_as_float((o >= 0) ? o : (o ^ 0x7fffffff));
}

// per-edge score + ordered-int atomicMax per destination node
__global__ void edge_phase1(const int* __restrict__ ei,
                            const float* __restrict__ asrc, const float* __restrict__ adst,
                            float* __restrict__ score, int* __restrict__ mbuf, int Ecnt) {
    int e = blockIdx.x * blockDim.x + threadIdx.x;
    if (e >= Ecnt) return;
    int s = ei[e], d = ei[Ecnt + e];
#pragma unroll
    for (int h = 0; h < HEADS; ++h) {
        float v = asrc[s * HEADS + h] + adst[d * HEADS + h];
        v = (v > 0.f) ? v : 0.2f * v;          // leaky_relu(0.2)
        score[(size_t)e * HEADS + h] = v;
        atomicMax(&mbuf[d * HEADS + h], f2ord(v));
    }
}
// exp(score - max) + denominator atomicAdd
__global__ void edge_phase2(const int* __restrict__ ei,
                            float* __restrict__ score, const int* __restrict__ mbuf,
                            float* __restrict__ denom, int Ecnt) {
    int e = blockIdx.x * blockDim.x + threadIdx.x;
    if (e >= Ecnt) return;
    int d = ei[Ecnt + e];
#pragma unroll
    for (int h = 0; h < HEADS; ++h) {
        float mx = ord2f(mbuf[d * HEADS + h]);
        float ex = __expf(score[(size_t)e * HEADS + h] - mx);
        score[(size_t)e * HEADS + h] = ex;
        atomicAdd(&denom[d * HEADS + h], ex);
    }
}
// wave per edge: each lane scatters 4 of the 128 message channels
__global__ void edge_phase3(const int* __restrict__ ei,
                            const float* __restrict__ score, const float* __restrict__ denom,
                            const float* __restrict__ xp, float* __restrict__ outacc, int Ecnt) {
    int gt = blockIdx.x * blockDim.x + threadIdx.x;
    int e = gt >> 5;
    if (e >= Ecnt) return;
    int lane = gt & 31;
    int s = ei[e], d = ei[Ecnt + e];
    int cbase = lane * 4;                       // 32 lanes x 4 = 128 channels
    int h = cbase >> 6;                         // head for this channel block
    float alpha = score[(size_t)e * HEADS + h] / (denom[d * HEADS + h] + 1e-16f);
    const float4 xv = *reinterpret_cast<const float4*>(xp + (size_t)s * HID + cbase);
    float* op = outacc + (size_t)d * HID + cbase;
    atomicAdd(op + 0, xv.x * alpha);
    atomicAdd(op + 1, xv.y * alpha);
    atomicAdd(op + 2, xv.z * alpha);
    atomicAdd(op + 3, xv.w * alpha);
}

// bias(sum of both relations) + leaky_relu(0.01) + x2, emit f16 for next GEMMs
__global__ void post_gat(const float* __restrict__ acc, const float* __restrict__ bgat_l,
                         half_t* __restrict__ act16, int total) {
    int i = blockIdx.x * blockDim.x + threadIdx.x;
    if (i >= total) return;
    int c = i & (HID - 1);
    float v = acc[i] + bgat_l[c] + bgat_l[HID + c];
    v = (v > 0.f) ? v : 0.01f * v;
    act16[i] = (half_t)(2.f * v);
}

// ---------------------------------------------------------------------------
// host orchestration
// ---------------------------------------------------------------------------
static inline char* carve(char*& p, size_t bytes) {
    char* q = p;
    p += (bytes + 255) & ~(size_t)255;
    return q;
}

extern "C" void kernel_launch(void* const* d_in, const int* in_sizes, int n_in,
                              void* d_out, int out_size, void* d_ws, size_t ws_size,
                              hipStream_t stream) {
    const int N = in_sizes[0] / 64;      // 50000
    const int E = in_sizes[1] / 2;       // 600000
    const int IN_F = 64, L = 2, R = 2;

    const float* x        = (const float*)d_in[0];
    const int*   ei[2]    = { (const int*)d_in[1], (const int*)d_in[2] };
    const float* w_emb1   = (const float*)d_in[3];
    const float* g_emb    = (const float*)d_in[4];
    const float* b_emb    = (const float*)d_in[5];
    const float* w_emb2   = (const float*)d_in[6];
    const float* W_gat    = (const float*)d_in[7];
    const float* a_src    = (const float*)d_in[8];
    const float* a_dst    = (const float*)d_in[9];
    const float* b_gat    = (const float*)d_in[10];
    const float* w_dec1   = (const float*)d_in[11];
    const float* g_dec    = (const float*)d_in[12];
    const float* b_dec    = (const float*)d_in[13];
    const float* w_dec2   = (const float*)d_in[14];

    // ---- workspace layout ----
    char* p = (char*)d_ws;
    half_t* act_a  = (half_t*)carve(p, (size_t)N * HID * sizeof(half_t)); // also x16 (N x 64)
    half_t* act_b  = (half_t*)carve(p, (size_t)N * HID * sizeof(half_t));
    float*  f32x   = (float*) carve(p, (size_t)N * HID * sizeof(float));  // h1 / xp / mlp mid
    float*  f32acc = (float*) carve(p, (size_t)N * HID * sizeof(float));  // GAT accumulator
    float*  asrc   = (float*) carve(p, (size_t)N * HEADS * sizeof(float));
    float*  adst   = (float*) carve(p, (size_t)N * HEADS * sizeof(float));
    int*    mbuf   = (int*)   carve(p, (size_t)N * HEADS * sizeof(int));
    float*  denom  = (float*) carve(p, (size_t)N * HEADS * sizeof(float));
    float*  score  = (float*) carve(p, (size_t)E * HEADS * sizeof(float));
    float*  stat_sum   = (float*)carve(p, HID * sizeof(float));
    float*  stat_sumsq = (float*)carve(p, HID * sizeof(float));
    float*  bn_scale   = (float*)carve(p, HID * sizeof(float));
    float*  bn_shift   = (float*)carve(p, HID * sizeof(float));
    half_t* w16_emb1 = (half_t*)carve(p, (size_t)IN_F * HID * sizeof(half_t));
    half_t* w16_emb2 = (half_t*)carve(p, (size_t)HID * HID * sizeof(half_t));
    half_t* w16_gat  = (half_t*)carve(p, (size_t)L * R * HID * HID * sizeof(half_t));
    half_t* w16_dec1 = (half_t*)carve(p, (size_t)HID * HID * sizeof(half_t));
    half_t* w16_dec2 = (half_t*)carve(p, (size_t)HID * 2 * sizeof(half_t));

    auto cdiv = [](int a, int b) { return (a + b - 1) / b; };
    const int TB = 256;

    // ---- weight + input conversion to f16 ----
    f32_to_f16<<<cdiv(IN_F * HID, TB), TB, 0, stream>>>(w_emb1, w16_emb1, IN_F * HID);
    f32_to_f16<<<cdiv(HID * HID, TB), TB, 0, stream>>>(w_emb2, w16_emb2, HID * HID);
    f32_to_f16<<<cdiv(L * R * HID * HID, TB), TB, 0, stream>>>(W_gat, w16_gat, L * R * HID * HID);
    f32_to_f16<<<cdiv(HID * HID, TB), TB, 0, stream>>>(w_dec1, w16_dec1, HID * HID);
    f32_to_f16<<<cdiv(HID * 2, TB), TB, 0, stream>>>(w_dec2, w16_dec2, HID * 2);
    f32_to_f16<<<cdiv(N * IN_F, TB), TB, 0, stream>>>(x, act_a, N * IN_F);

    const int gemm_blocks = cdiv(N, 64);     // 64 rows per block
    const int nh = N * HID;

    auto run_bn = [&](const float* gamma, const float* beta, const float* X, half_t* out16) {
        fill_f32<<<cdiv(HID, TB), TB, 0, stream>>>(stat_sum, 0.f, HID);
        fill_f32<<<cdiv(HID, TB), TB, 0, stream>>>(stat_sumsq, 0.f, HID);
        col_stats<<<256, HID, 0, stream>>>(X, stat_sum, stat_sumsq, N);
        finalize_stats<<<1, HID, 0, stream>>>(stat_sum, stat_sumsq, gamma, beta,
                                              bn_scale, bn_shift, N);
        bn_relu_to_f16<<<cdiv(nh, TB), TB, 0, stream>>>(X, bn_scale, bn_shift, out16, nh);
    };

    // ---- embed MLP:  h = relu(BN(x @ w_emb1)) @ w_emb2 ----
    gemm_f16_wmma<64, HID><<<gemm_blocks, TB, 0, stream>>>(act_a, w16_emb1, f32x, N);
    run_bn(g_emb, b_emb, f32x, act_b);
    gemm_f16_wmma<HID, HID><<<gemm_blocks, TB, 0, stream>>>(act_b, w16_emb2, f32x, N);
    f32_to_f16<<<cdiv(nh, TB), TB, 0, stream>>>(f32x, act_a, nh);   // h (f16) in act_a

    // ---- GAT layers ----
    for (int l = 0; l < L; ++l) {
        fill_f32<<<cdiv(nh, TB), TB, 0, stream>>>(f32acc, 0.f, nh);
        for (int r = 0; r < R; ++r) {
            const int lr = l * R + r;
            gemm_f16_wmma<HID, HID><<<gemm_blocks, TB, 0, stream>>>(
                act_a, w16_gat + (size_t)lr * HID * HID, f32x, N);
            gat_alphas<<<cdiv(N, TB), TB, 0, stream>>>(
                f32x, a_src + lr * HEADS * CHAN, a_dst + lr * HEADS * CHAN, asrc, adst, N);
            fill_i32<<<cdiv(N * HEADS, TB), TB, 0, stream>>>(mbuf, (int)0x80000000, N * HEADS);
            fill_f32<<<cdiv(N * HEADS, TB), TB, 0, stream>>>(denom, 0.f, N * HEADS);
            edge_phase1<<<cdiv(E, TB), TB, 0, stream>>>(ei[r], asrc, adst, score, mbuf, E);
            edge_phase2<<<cdiv(E, TB), TB, 0, stream>>>(ei[r], score, mbuf, denom, E);
            edge_phase3<<<cdiv(E * 32, TB), TB, 0, stream>>>(ei[r], score, denom, f32x, f32acc, E);
        }
        post_gat<<<cdiv(nh, TB), TB, 0, stream>>>(f32acc, b_gat + l * R * HID, act_a, nh);
    }

    // ---- decision MLP:  out = relu(BN(h @ w_dec1)) @ w_dec2  (N x 2, fp32) ----
    gemm_f16_wmma<HID, HID><<<gemm_blocks, TB, 0, stream>>>(act_a, w16_dec1, f32x, N);
    run_bn(g_dec, b_dec, f32x, act_b);
    gemm_f16_wmma<HID, 2><<<gemm_blocks, TB, 0, stream>>>(act_b, w16_dec2, (float*)d_out, N);
}
